// LSTM_62234076119550
// MI455X (gfx1250) — compile-verified, matrix operand order checked by reading, not run on hardware
//
#include <hip/hip_runtime.h>

// ---------------------------------------------------------------------------
// 2-layer LSTM for MI455X (gfx1250): bf16 WMMA GEMMs + persistent scan kernel
//  - w_hh B-fragments held in VGPRs across all 1024 timesteps (128 VGPRs/lane)
//  - pre-activations stored (T, B, 4H) so each step reads one contiguous 256KB
//    slice; next slice prefetched with global_prefetch_b8 during compute
// ---------------------------------------------------------------------------

typedef __attribute__((ext_vector_type(16))) __bf16        v16bf;
typedef __attribute__((ext_vector_type(8)))  float         v8f;
typedef __attribute__((ext_vector_type(4)))  unsigned int  u32x4;

union ABFrag { v16bf v; u32x4 u[2]; };

#define B_  64
#define T_  1024
#define I_  128
#define H_  512
#define G4_ 2048          /* 4*H */
#define O_  128
#define M_  (B_ * T_)     /* 65536 */

__device__ __forceinline__ v8f wmma_bf16(v16bf a, v16bf b, v8f c) {
    // D = A(16x32 bf16) x B(32x16 bf16) + C(16x16 f32)
    return __builtin_amdgcn_wmma_f32_16x16x32_bf16(
        /*neg_a=*/false, a, /*neg_b=*/false, b,
        /*c_mod=*/(short)0, c, /*reuse_a=*/false, /*reuse_b=*/false);
}

__device__ __forceinline__ float sigmoidf_(float x) {
    return 1.0f / (1.0f + __expf(-x));
}
__device__ __forceinline__ float tanhf_(float x) {
    float e = __expf(2.0f * x);
    return 1.0f - 2.0f / (e + 1.0f);   // safe at +/-inf
}

// ---------------------------------------------------------------------------
// fp32 -> bf16 conversion
// ---------------------------------------------------------------------------
__global__ void f2bf_kernel(const float* __restrict__ src,
                            __bf16* __restrict__ dst, int n) {
    int i = blockIdx.x * blockDim.x + threadIdx.x;
    if (i < n) dst[i] = (__bf16)src[i];
}

// ---------------------------------------------------------------------------
// pre[m,n] = sum_k A[m,k]*W[n,k] + bias0[n] + bias1[n]   (bf16 out)
// A: (M,K) bf16 row-major (m = b*T + t), W: (N,K) bf16 row-major.
// transT != 0: store transposed as out[(t*B + b)*N + n] for the scan kernel.
// Block = 8 waves, each wave one 16x16 tile; block tile = 32(M) x 64(N).
// grid = (N/64, M/32). K % 64 == 0.
// ---------------------------------------------------------------------------
__global__ void gemm_pre(const __bf16* __restrict__ A, const __bf16* __restrict__ W,
                         const float* __restrict__ bias0, const float* __restrict__ bias1,
                         __bf16* __restrict__ out, int K, int N, int transT) {
    int wv   = threadIdx.x >> 5;
    int lane = threadIdx.x & 31;
    int tm   = blockIdx.y * 2 + (wv & 1);
    int tn   = blockIdx.x * 4 + (wv >> 1);
    int lm   = lane & 15;
    int ksel = (lane >> 4) << 3;          // 0 or 8: K-half selector (ISA A/B layout)

    const __bf16* arow = A + (long)(tm * 16 + lm) * K;
    const __bf16* wrow = W + (long)(tn * 16 + lm) * K;

    v8f acc0 = {}; v8f acc1 = {};
    for (int k = 0; k < K; k += 64) {
        ABFrag a0, b0, a1, b1;
        int k0 = k + ksel;
        a0.u[0] = *(const u32x4*)(arow + k0);
        a0.u[1] = *(const u32x4*)(arow + k0 + 16);
        b0.u[0] = *(const u32x4*)(wrow + k0);
        b0.u[1] = *(const u32x4*)(wrow + k0 + 16);
        a1.u[0] = *(const u32x4*)(arow + k0 + 32);
        a1.u[1] = *(const u32x4*)(arow + k0 + 48);
        b1.u[0] = *(const u32x4*)(wrow + k0 + 32);
        b1.u[1] = *(const u32x4*)(wrow + k0 + 48);
        acc0 = wmma_bf16(a0.v, b0.v, acc0);
        acc1 = wmma_bf16(a1.v, b1.v, acc1);
    }

    int n     = tn * 16 + lm;
    int rbase = tm * 16 + ksel;           // C/D rows: r + 8*(lane>=16)
    float bias = bias0[n] + bias1[n];
#pragma unroll
    for (int r = 0; r < 8; ++r) {
        float v = acc0[r] + acc1[r] + bias;
        long m = rbase + r;               // m = b*T + t
        long oidx = transT ? (((m & (T_ - 1)) * B_ + (m >> 10)) * (long)N + n)
                           : (m * (long)N + n);
        out[oidx] = (__bf16)v;
    }
}

// ---------------------------------------------------------------------------
// FC: out[m*129 + n] = sum_k A[m,k]*W[n,k] + bias[n]   (fp32 out, ldc=129)
// grid = (O/64, M/32)
// ---------------------------------------------------------------------------
__global__ void gemm_fc(const __bf16* __restrict__ A, const __bf16* __restrict__ W,
                        const float* __restrict__ bias, float* __restrict__ out, int K) {
    int wv   = threadIdx.x >> 5;
    int lane = threadIdx.x & 31;
    int tm   = blockIdx.y * 2 + (wv & 1);
    int tn   = blockIdx.x * 4 + (wv >> 1);
    int lm   = lane & 15;
    int ksel = (lane >> 4) << 3;

    const __bf16* arow = A + (long)(tm * 16 + lm) * K;
    const __bf16* wrow = W + (long)(tn * 16 + lm) * K;

    v8f acc0 = {}; v8f acc1 = {};
    for (int k = 0; k < K; k += 64) {
        ABFrag a0, b0, a1, b1;
        int k0 = k + ksel;
        a0.u[0] = *(const u32x4*)(arow + k0);
        a0.u[1] = *(const u32x4*)(arow + k0 + 16);
        b0.u[0] = *(const u32x4*)(wrow + k0);
        b0.u[1] = *(const u32x4*)(wrow + k0 + 16);
        a1.u[0] = *(const u32x4*)(arow + k0 + 32);
        a1.u[1] = *(const u32x4*)(arow + k0 + 48);
        b1.u[0] = *(const u32x4*)(wrow + k0 + 32);
        b1.u[1] = *(const u32x4*)(wrow + k0 + 48);
        acc0 = wmma_bf16(a0.v, b0.v, acc0);
        acc1 = wmma_bf16(a1.v, b1.v, acc1);
    }

    int n     = tn * 16 + lm;
    int rbase = tm * 16 + ksel;
    float bv  = bias[n];
#pragma unroll
    for (int r = 0; r < 8; ++r)
        out[(long)(rbase + r) * 129 + n] = acc0[r] + acc1[r] + bv;
}

// ---------------------------------------------------------------------------
// software grid barrier (all 64 blocks co-resident): bar[0]=count, bar[1]=gen
// ---------------------------------------------------------------------------
__device__ __forceinline__ void grid_sync(int* bar, int nblk) {
    __threadfence();
    __syncthreads();
    if (threadIdx.x == 0) {
        int g = __hip_atomic_load(&bar[1], __ATOMIC_RELAXED, __HIP_MEMORY_SCOPE_AGENT);
        if (__hip_atomic_fetch_add(&bar[0], 1, __ATOMIC_ACQ_REL,
                                   __HIP_MEMORY_SCOPE_AGENT) == nblk - 1) {
            __hip_atomic_store(&bar[0], 0, __ATOMIC_RELAXED, __HIP_MEMORY_SCOPE_AGENT);
            __threadfence();
            __hip_atomic_fetch_add(&bar[1], 1, __ATOMIC_RELEASE, __HIP_MEMORY_SCOPE_AGENT);
        } else {
            while (__hip_atomic_load(&bar[1], __ATOMIC_ACQUIRE,
                                     __HIP_MEMORY_SCOPE_AGENT) == g) {
                __builtin_amdgcn_s_sleep(2);
            }
        }
    }
    __syncthreads();
}

// ---------------------------------------------------------------------------
// Persistent LSTM scan: 1024 steps of
//   gates = pre[t,:,:] + h @ whh^T ; (i,f,g,o) nonlinearity ; c,h update
// 64 blocks x 256 threads = 512 waves; wave -> one 16x16 tile of (64 x 2048).
// pre:  (T, B, 4H) bf16 (contiguous 256KB per step)
// whh:  (4H, H)   bf16 row-major; the wave's 16x512 B-tile lives in VGPRs
// hbuf: (B, H) bf16 current hidden; cbuf: (B, H) f32; gates: (B, 4H) f32
// hseq: (B*T, H) bf16 output sequence (m = b*T + t)
// ---------------------------------------------------------------------------
__global__ void __launch_bounds__(256, 1)
lstm_scan(const __bf16* __restrict__ pre, const __bf16* __restrict__ whh,
          __bf16* __restrict__ hbuf, float* __restrict__ cbuf,
          float* __restrict__ gates, __bf16* __restrict__ hseq,
          int* __restrict__ bar) {
    const int nblk = gridDim.x;
    int tid  = threadIdx.x;
    int gtid = blockIdx.x * 256 + tid;     // 0..16383
    int wv   = gtid >> 5;                  // 0..511
    int lane = tid & 31;
    int tm   = wv & 3;                     // M tile (batch rows)
    int tn   = wv >> 2;                    // N tile (gate cols)
    int lm   = lane & 15;
    int ksel = (lane >> 4) << 3;

    int n     = tn * 16 + lm;
    int rbase = tm * 16 + ksel;
    const __bf16* arow = hbuf + (tm * 16 + lm) * H_;
    const __bf16* wrow = whh + (long)(tn * 16 + lm) * H_;

    // ---- hoist the wave's whole B-tile (16x512 bf16) into VGPRs ----
    ABFrag bfr[16];
#pragma unroll
    for (int kk = 0; kk < 16; ++kk) {
        int k0 = kk * 32 + ksel;
        bfr[kk].u[0] = *(const u32x4*)(wrow + k0);
        bfr[kk].u[1] = *(const u32x4*)(wrow + k0 + 16);
    }

    for (int t = 0; t < T_; ++t) {
        const __bf16* preslice = pre + (long)t * (B_ * G4_);

        // ---- GEMM phase: accumulate onto pre-activations ----
        v8f acc0 = {}; v8f acc1 = {};
#pragma unroll
        for (int r = 0; r < 8; ++r)
            acc0[r] = (float)preslice[(rbase + r) * G4_ + n];

        // prefetch next step's pre slice while this step computes
        if (t + 1 < T_) {
            const __bf16* nxt = preslice + B_ * G4_;
#pragma unroll
            for (int r = 0; r < 8; r += 2)
                __builtin_prefetch(nxt + (rbase + r) * G4_ + n, 0, 0);
        }

#pragma unroll
        for (int kk = 0; kk < 16; kk += 2) {
            ABFrag a0, a1;
            int k0 = kk * 32 + ksel;
            a0.u[0] = *(const u32x4*)(arow + k0);
            a0.u[1] = *(const u32x4*)(arow + k0 + 16);
            a1.u[0] = *(const u32x4*)(arow + k0 + 32);
            a1.u[1] = *(const u32x4*)(arow + k0 + 48);
            acc0 = wmma_bf16(a0.v, bfr[kk].v, acc0);
            acc1 = wmma_bf16(a1.v, bfr[kk + 1].v, acc1);
        }
#pragma unroll
        for (int r = 0; r < 8; ++r)
            gates[(rbase + r) * G4_ + n] = acc0[r] + acc1[r];

        grid_sync(bar, nblk);

        // ---- elementwise phase: 64*512 cells over 16384 threads ----
#pragma unroll
        for (int rep = 0; rep < 2; ++rep) {
            int j = gtid + rep * 16384;    // 0..32767
            int b = j >> 9, u = j & 511;
            const float* grow = gates + b * G4_;
            float ig = sigmoidf_(grow[u]);
            float fg = sigmoidf_(grow[H_ + u]);
            float gg = tanhf_(grow[2 * H_ + u]);
            float og = sigmoidf_(grow[3 * H_ + u]);
            float c  = fg * cbuf[j] + ig * gg;
            float h  = og * tanhf_(c);
            cbuf[j] = c;
            __bf16 hb = (__bf16)h;
            hbuf[j] = hb;
            hseq[((long)b * T_ + t) * H_ + u] = hb;
        }
        grid_sync(bar, nblk);
    }
}

// out[m*129 + 128] = x[m*128 + 127]  (the "pitch" concat column)
__global__ void pitch_kernel(const float* __restrict__ x, float* __restrict__ out) {
    int m = blockIdx.x * blockDim.x + threadIdx.x;
    out[(long)m * 129 + 128] = x[(long)m * I_ + (I_ - 1)];
}

// ---------------------------------------------------------------------------
extern "C" void kernel_launch(void* const* d_in, const int* in_sizes, int n_in,
                              void* d_out, int out_size, void* d_ws, size_t ws_size,
                              hipStream_t stream) {
    const float* x     = (const float*)d_in[0];
    const float* wih0f = (const float*)d_in[1];
    const float* whh0f = (const float*)d_in[2];
    const float* bih0  = (const float*)d_in[3];
    const float* bhh0  = (const float*)d_in[4];
    const float* wih1f = (const float*)d_in[5];
    const float* whh1f = (const float*)d_in[6];
    const float* bih1  = (const float*)d_in[7];
    const float* bhh1  = (const float*)d_in[8];
    const float* fcwf  = (const float*)d_in[9];
    const float* fcb   = (const float*)d_in[10];
    float* out = (float*)d_out;

    char* ws = (char*)d_ws;
    size_t off = 0;
    auto alloc = [&](size_t bytes) -> void* {
        void* p = ws + off;
        off = (off + bytes + 255) & ~(size_t)255;
        return p;
    };

    int*    bar   = (int*)   alloc(256);
    __bf16* xbf   = (__bf16*)alloc((size_t)M_ * I_ * 2);
    __bf16* wih0  = (__bf16*)alloc((size_t)G4_ * I_ * 2);
    __bf16* whh0  = (__bf16*)alloc((size_t)G4_ * H_ * 2);
    __bf16* wih1  = (__bf16*)alloc((size_t)G4_ * H_ * 2);
    __bf16* whh1  = (__bf16*)alloc((size_t)G4_ * H_ * 2);
    __bf16* fcw   = (__bf16*)alloc((size_t)O_ * H_ * 2);
    __bf16* pre   = (__bf16*)alloc((size_t)M_ * G4_ * 2);   // reused by both layers
    __bf16* hs0   = (__bf16*)alloc((size_t)M_ * H_ * 2);
    __bf16* hs1   = (__bf16*)alloc((size_t)M_ * H_ * 2);
    float*  gates = (float*) alloc((size_t)B_ * G4_ * 4);
    float*  cbuf  = (float*) alloc((size_t)B_ * H_ * 4);
    __bf16* hbuf  = (__bf16*)alloc((size_t)B_ * H_ * 2);

    auto cvt = [&](const float* s, __bf16* d, int n) {
        f2bf_kernel<<<(n + 255) / 256, 256, 0, stream>>>(s, d, n);
    };
    cvt(x,     xbf,  M_ * I_);
    cvt(wih0f, wih0, G4_ * I_);
    cvt(whh0f, whh0, G4_ * H_);
    cvt(wih1f, wih1, G4_ * H_);
    cvt(whh1f, whh1, G4_ * H_);
    cvt(fcwf,  fcw,  O_ * H_);

    // ---- layer 0 ----
    gemm_pre<<<dim3(G4_ / 64, M_ / 32), 256, 0, stream>>>(
        xbf, wih0, bih0, bhh0, pre, I_, G4_, /*transT=*/1);
    hipMemsetAsync(bar,  0, 256,            stream);
    hipMemsetAsync(cbuf, 0, B_ * H_ * 4,    stream);
    hipMemsetAsync(hbuf, 0, B_ * H_ * 2,    stream);
    lstm_scan<<<64, 256, 0, stream>>>(pre, whh0, hbuf, cbuf, gates, hs0, bar);

    // ---- layer 1 ----
    gemm_pre<<<dim3(G4_ / 64, M_ / 32), 256, 0, stream>>>(
        hs0, wih1, bih1, bhh1, pre, H_, G4_, /*transT=*/1);
    hipMemsetAsync(bar,  0, 256,            stream);
    hipMemsetAsync(cbuf, 0, B_ * H_ * 4,    stream);
    hipMemsetAsync(hbuf, 0, B_ * H_ * 2,    stream);
    lstm_scan<<<64, 256, 0, stream>>>(pre, whh1, hbuf, cbuf, gates, hs1, bar);

    // ---- FC + pitch concat ----
    gemm_fc<<<dim3(O_ / 64, M_ / 32), 256, 0, stream>>>(hs1, fcw, fcb, out, H_);
    pitch_kernel<<<M_ / 256, 256, 0, stream>>>(x, out);
}
